// Local_ViT_44487271252027
// MI455X (gfx1250) — matrix operand order repaired and checked by reading, not compiled, not run on hardware
//
#include <hip/hip_runtime.h>
#include <math.h>

// ---------------------------------------------------------------------------
// Fused Swin-style local ViT block for MI455X (gfx1250, wave32, WMMA).
// One workgroup per 8x8 window (4096 windows). Whole block stays in LDS:
// LN1 -> QKV -> attn(+rel bias, softmax) -> proj -> +res -> LN2 -> MLP -> +res
// All GEMMs use v_wmma_f32_16x16x32_bf16 with f32 accumulate.
// Round 3: wave id pinned to SGPR via readfirstlane so every tile loop and
// the QKV q/k/v branch is scalar (s_cbranch) instead of exec-masked.
// ---------------------------------------------------------------------------

#define B_      4
#define H_      256
#define W_      256
#define C_      96
#define WIN_    8
#define HEADS_  3
#define HD_     32
#define N_      64          // tokens per window
#define HID_    384
#define NWIN    4096        // B * (H/WIN) * (W/WIN)
#define NTHREADS 256
#define SCALE_  0.1767766952966369f   // 32^-0.5

typedef __bf16 bf16_t;
typedef bf16_t v16bf __attribute__((ext_vector_type(16)));
typedef bf16_t v8bf  __attribute__((ext_vector_type(8)));
typedef float  v8f   __attribute__((ext_vector_type(8)));
typedef float  v4f   __attribute__((ext_vector_type(4)));

// ---- LDS working set (~236 KB < 320 KB/WGP) -------------------------------
struct SMem {
  bf16_t wbuf[HID_ * C_];            // 73728B: current weight, transposed [n][k] bf16
  float  xwin[N_ * C_];              // 24576B: window activations f32 (becomes x2)
  bf16_t hbuf[N_ * C_];              // 12288B: LN1/LN2 output (A matrix)
  bf16_t qs[HEADS_ * N_ * HD_];      // 12288B: q * scale, [h][m][d]
  bf16_t ks[HEADS_ * N_ * HD_];      // 12288B: k, [h][j][d]  (serves as B matrix)
  bf16_t vt[HEADS_ * HD_ * N_];      // 12288B: v^T, [h][d][j] (serves as B matrix)
  bf16_t probs[HEADS_ * N_ * N_];    // 24576B: softmax probs (A matrix)
  bf16_t aout[N_ * C_];              // 12288B: attention output (A matrix for proj)
  union {
    float  scores[HEADS_ * N_ * N_]; // 49152B: pre-softmax logits f32
    bf16_t mlph[N_ * HID_];          // 49152B: gelu(fc1) (A matrix for fc2)
  } u;
  float bq[3 * C_], bp[C_], bf1[HID_], bf2[C_];
  float g1[C_], b1[C_], g2[C_], b2[C_];
  float relt[(2 * WIN_ - 1) * (2 * WIN_ - 1) * HEADS_]; // 675 f32
};

// ---- WMMA bf16 fragment loaders (CDNA5 16x16x32 layouts, wave32) ----------
// A 16x32 (MxK): lane L<16 -> M=L, K in {half*8..+7} U {16+half*8..+7}; half=lane>>4
__device__ __forceinline__ v16bf load_a_frag(const bf16_t* A, int lda,
                                             int m0, int kb, int lane) {
  const int half = lane >> 4;
  const bf16_t* p = A + (m0 + (lane & 15)) * lda + kb + half * 8;
  v8bf lo = *(const v8bf*)p;
  v8bf hi = *(const v8bf*)(p + 16);
  return __builtin_shufflevector(lo, hi, 0, 1, 2, 3, 4, 5, 6, 7,
                                 8, 9, 10, 11, 12, 13, 14, 15);
}

// B 32x16 (KxN), stored transposed [n][k]: lane -> N=lane&15, K in {half*8..+7} U {16+half*8..+7}
__device__ __forceinline__ v16bf load_b_frag(const bf16_t* Bt, int ldb,
                                             int n0, int kb, int lane) {
  const int half = lane >> 4;
  const bf16_t* p = Bt + (n0 + (lane & 15)) * ldb + kb + half * 8;
  v8bf lo = *(const v8bf*)p;
  v8bf hi = *(const v8bf*)(p + 16);
  return __builtin_shufflevector(lo, hi, 0, 1, 2, 3, 4, 5, 6, 7,
                                 8, 9, 10, 11, 12, 13, 14, 15);
}

// 16x16 output tile, accumulating over K (multiple of 32).
// C/D element e maps to (m = m0 + (lane>>4)*8 + e, n = n0 + (lane&15)).
__device__ __forceinline__ v8f mm_tile(const bf16_t* A, int lda,
                                       const bf16_t* Bt, int ldb,
                                       int m0, int n0, int K, int lane) {
  v8f acc = {};
  for (int kb = 0; kb < K; kb += 32) {
    v16bf a = load_a_frag(A, lda, m0, kb, lane);
    v16bf b = load_b_frag(Bt, ldb, n0, kb, lane);
    acc = __builtin_amdgcn_wmma_f32_16x16x32_bf16(
        false, a, false, b, (short)0, acc, false, false);
  }
  return acc;
}

__global__ __launch_bounds__(NTHREADS, 1)
void swin_block_kernel(const float* __restrict__ x,
                       const float* __restrict__ n1g, const float* __restrict__ n1b,
                       const float* __restrict__ qkv_w, const float* __restrict__ qkv_b,
                       const float* __restrict__ proj_w, const float* __restrict__ proj_b,
                       const float* __restrict__ relb,
                       const float* __restrict__ n2g, const float* __restrict__ n2b,
                       const float* __restrict__ fc1_w, const float* __restrict__ fc1_b,
                       const float* __restrict__ fc2_w, const float* __restrict__ fc2_b,
                       float* __restrict__ out) {
  __shared__ SMem sm;
  const int tid  = threadIdx.x;
  // wave id is lane-invariant: pin it to an SGPR so all tile indices and
  // stage branches become scalar control flow (s_cmp/s_cbranch, no exec mask)
  const int wid  = __builtin_amdgcn_readfirstlane(tid >> 5);
  const int lane = tid & 31;
  const int win  = blockIdx.x;
  const int b    = win >> 10;
  const int wy   = (win >> 5) & 31;
  const int wx   = win & 31;

  auto gidx = [&](int m) {
    int y = (wy << 3) + (m >> 3);
    int xc = (wx << 3) + (m & 7);
    return ((b * H_ + y) * W_ + xc) * C_;
  };

  // ---- stage 0: window acts (vectorized), params, qkv weights -------------
  if (tid == 0) __builtin_prefetch(qkv_w, 0, 3);  // global_prefetch_b8
  // window row r holds tokens 8r..8r+7 contiguously: 768 floats = 192 float4
  for (int idx = tid; idx < 8 * 192; idx += NTHREADS) {
    int r = idx / 192, o = idx - r * 192;
    const v4f* src = (const v4f*)(x + ((b * H_ + (wy << 3) + r) * W_ + (wx << 3)) * C_) + o;
    *((v4f*)(sm.xwin + r * 768) + o) = *src;
  }
  for (int i = tid; i < 675; i += NTHREADS) sm.relt[i] = relb[i];
  for (int i = tid; i < 3 * C_; i += NTHREADS) sm.bq[i] = qkv_b[i];
  for (int i = tid; i < HID_; i += NTHREADS) sm.bf1[i] = fc1_b[i];
  for (int i = tid; i < C_; i += NTHREADS) {
    sm.bp[i] = proj_b[i]; sm.bf2[i] = fc2_b[i];
    sm.g1[i] = n1g[i]; sm.b1[i] = n1b[i];
    sm.g2[i] = n2g[i]; sm.b2[i] = n2b[i];
  }
  for (int idx = tid; idx < C_ * 3 * C_; idx += NTHREADS) {  // qkv_w [k][n] -> wbuf [n][k]
    int k = idx / (3 * C_), n = idx - k * (3 * C_);
    sm.wbuf[n * C_ + k] = (bf16_t)qkv_w[idx];
  }
  __syncthreads();

  // ---- LN1 ----------------------------------------------------------------
  if (tid < N_) {
    const float* row = &sm.xwin[tid * C_];
    float mu = 0.f;
    for (int c = 0; c < C_; ++c) mu += row[c];
    mu *= (1.f / C_);
    float var = 0.f;
    for (int c = 0; c < C_; ++c) { float d = row[c] - mu; var += d * d; }
    var *= (1.f / C_);
    float rs = rsqrtf(var + 1e-5f);
    for (int c = 0; c < C_; ++c)
      sm.hbuf[tid * C_ + c] = (bf16_t)((row[c] - mu) * rs * sm.g1[c] + sm.b1[c]);
  }
  __syncthreads();

  // ---- QKV: (64x96) @ (96x288), split into q*scale / k / v^T --------------
  // Column-section indices are TILE-uniform (scalar): s = ni/6, h = (ni%6)/2.
  for (int t = wid; t < 72; t += 8) {           // 4 x 18 tiles
    int mi = t / 18, ni = t - (t / 18) * 18;
    v8f acc = mm_tile(sm.hbuf, C_, sm.wbuf, C_, mi * 16, ni * 16, C_, lane);
    int s   = ni / 6;                 // 0=q, 1=k, 2=v   (scalar)
    int rem = ni - s * 6;
    int h   = rem >> 1;               // head            (scalar)
    int d   = ((rem & 1) << 4) + (lane & 15);
    float bias = sm.bq[ni * 16 + (lane & 15)];
    int mb = mi * 16 + ((lane >> 4) << 3);
    if (s == 0) {
#pragma unroll
      for (int e = 0; e < 8; ++e)
        sm.qs[(h * N_ + mb + e) * HD_ + d] = (bf16_t)((acc[e] + bias) * SCALE_);
    } else if (s == 1) {
#pragma unroll
      for (int e = 0; e < 8; ++e)
        sm.ks[(h * N_ + mb + e) * HD_ + d] = (bf16_t)(acc[e] + bias);
    } else {
      v8bf pk;                        // consecutive e -> consecutive m: 1x b128 store
#pragma unroll
      for (int e = 0; e < 8; ++e) pk[e] = (bf16_t)(acc[e] + bias);
      *(v8bf*)&sm.vt[(h * HD_ + d) * N_ + mb] = pk;
    }
  }
  __syncthreads();

  // restage proj_w^T while computing scores (wbuf free until proj stage)
  for (int idx = tid; idx < C_ * C_; idx += NTHREADS) {
    int k = idx / C_, n = idx - k * C_;
    sm.wbuf[n * C_ + k] = (bf16_t)proj_w[idx];
  }
  // ---- attention logits: q k^T + rel-pos bias -----------------------------
  for (int t = wid; t < 48; t += 8) {           // 3 heads x 4x4 tiles
    int h = t / 16, r = t - h * 16, mi = r >> 2, nj = r & 3;
    v8f acc = mm_tile(&sm.qs[h * N_ * HD_], HD_,
                      &sm.ks[h * N_ * HD_], HD_, mi * 16, nj * 16, HD_, lane);
    int j = nj * 16 + (lane & 15);
    int jy = j >> 3, jx = j & 7;
    int ib = mi * 16 + ((lane >> 4) << 3);
#pragma unroll
    for (int e = 0; e < 8; ++e) {
      int i = ib + e;
      int dy = (i >> 3) - jy + 7, dx = (i & 7) - jx + 7;
      sm.u.scores[(h * N_ + i) * N_ + j] =
          acc[e] + sm.relt[(dy * 15 + dx) * HEADS_ + h];
    }
  }
  __syncthreads();

  // ---- softmax (one thread per (head,row)) --------------------------------
  if (tid < HEADS_ * N_) {
    float* row = &sm.u.scores[tid * N_];
    float mx = -1e30f;
    for (int j = 0; j < N_; ++j) mx = fmaxf(mx, row[j]);
    float ssum = 0.f;
    for (int j = 0; j < N_; ++j) { float e = expf(row[j] - mx); row[j] = e; ssum += e; }
    float inv = 1.f / ssum;
    for (int j8 = 0; j8 < N_; j8 += 8) {        // packed b128 stores
      v8bf pk;
#pragma unroll
      for (int e = 0; e < 8; ++e) pk[e] = (bf16_t)(row[j8 + e] * inv);
      *(v8bf*)&sm.probs[tid * N_ + j8] = pk;
    }
  }
  __syncthreads();

  // ---- attn @ v -----------------------------------------------------------
  for (int t = wid; t < 24; t += 8) {           // 3 heads x 4x2 tiles
    int h = t >> 3, r = t & 7, mi = r >> 1, dj = r & 1;
    v8f acc = mm_tile(&sm.probs[h * N_ * N_], N_,
                      &sm.vt[h * HD_ * N_], N_, mi * 16, dj * 16, N_, lane);
    int d = dj * 16 + (lane & 15);
    int mb = mi * 16 + ((lane >> 4) << 3);
#pragma unroll
    for (int e = 0; e < 8; ++e)
      sm.aout[(mb + e) * C_ + h * HD_ + d] = (bf16_t)acc[e];
  }
  __syncthreads();

  // ---- proj + residual into xwin (becomes x2) -----------------------------
  for (int t = wid; t < 24; t += 8) {           // 4 x 6 tiles
    int mi = t / 6, ni = t - (t / 6) * 6;
    v8f acc = mm_tile(sm.aout, C_, sm.wbuf, C_, mi * 16, ni * 16, C_, lane);
    int n = ni * 16 + (lane & 15);
    float bias = sm.bp[n];
    int mb = mi * 16 + ((lane >> 4) << 3);
#pragma unroll
    for (int e = 0; e < 8; ++e) {
      int m = mb + e;
      sm.xwin[m * C_ + n] += acc[e] + bias;
    }
  }
  __syncthreads();

  // restage fc1_w^T; LN2 on x2 -> hbuf
  for (int idx = tid; idx < C_ * HID_; idx += NTHREADS) {
    int k = idx / HID_, n = idx - k * HID_;
    sm.wbuf[n * C_ + k] = (bf16_t)fc1_w[idx];
  }
  if (tid < N_) {
    const float* row = &sm.xwin[tid * C_];
    float mu = 0.f;
    for (int c = 0; c < C_; ++c) mu += row[c];
    mu *= (1.f / C_);
    float var = 0.f;
    for (int c = 0; c < C_; ++c) { float d = row[c] - mu; var += d * d; }
    var *= (1.f / C_);
    float rs = rsqrtf(var + 1e-5f);
    for (int c = 0; c < C_; ++c)
      sm.hbuf[tid * C_ + c] = (bf16_t)((row[c] - mu) * rs * sm.g2[c] + sm.b2[c]);
  }
  __syncthreads();

  // ---- fc1 + exact GELU ---------------------------------------------------
  for (int t = wid; t < 96; t += 8) {           // 4 x 24 tiles
    int mi = t / 24, ni = t - (t / 24) * 24;
    v8f acc = mm_tile(sm.hbuf, C_, sm.wbuf, C_, mi * 16, ni * 16, C_, lane);
    int n = ni * 16 + (lane & 15);
    float bias = sm.bf1[n];
    int mb = mi * 16 + ((lane >> 4) << 3);
#pragma unroll
    for (int e = 0; e < 8; ++e) {
      float v = acc[e] + bias;
      v = 0.5f * v * (1.f + erff(v * 0.70710678118654752f));
      sm.u.mlph[(mb + e) * HID_ + n] = (bf16_t)v;
    }
  }
  __syncthreads();

  // restage fc2_w^T: (384,96) [k][n] -> wbuf [n*384 + k]
  for (int idx = tid; idx < HID_ * C_; idx += NTHREADS) {
    int k = idx / C_, n = idx - k * C_;
    sm.wbuf[n * HID_ + k] = (bf16_t)fc2_w[idx];
  }
  __syncthreads();

  // ---- fc2 + final residual, write out ------------------------------------
  for (int t = wid; t < 24; t += 8) {           // 4 x 6 tiles, K = 384
    int mi = t / 6, ni = t - (t / 6) * 6;
    v8f acc = mm_tile(sm.u.mlph, HID_, sm.wbuf, HID_, mi * 16, ni * 16, HID_, lane);
    int n = ni * 16 + (lane & 15);
    float bias = sm.bf2[n];
    int mb = mi * 16 + ((lane >> 4) << 3);
#pragma unroll
    for (int e = 0; e < 8; ++e) {
      int m = mb + e;
      out[gidx(m) + n] = sm.xwin[m * C_ + n] + acc[e] + bias;
    }
  }
}

extern "C" void kernel_launch(void* const* d_in, const int* in_sizes, int n_in,
                              void* d_out, int out_size, void* d_ws, size_t ws_size,
                              hipStream_t stream) {
  (void)in_sizes; (void)n_in; (void)out_size; (void)d_ws; (void)ws_size;
  const float* x      = (const float*)d_in[0];
  const float* n1g    = (const float*)d_in[1];
  const float* n1b    = (const float*)d_in[2];
  const float* qkv_w  = (const float*)d_in[3];
  const float* qkv_b  = (const float*)d_in[4];
  const float* proj_w = (const float*)d_in[5];
  const float* proj_b = (const float*)d_in[6];
  const float* relb   = (const float*)d_in[7];
  const float* n2g    = (const float*)d_in[8];
  const float* n2b    = (const float*)d_in[9];
  const float* fc1_w  = (const float*)d_in[10];
  const float* fc1_b  = (const float*)d_in[11];
  const float* fc2_w  = (const float*)d_in[12];
  const float* fc2_b  = (const float*)d_in[13];
  float* out = (float*)d_out;

  swin_block_kernel<<<NWIN, NTHREADS, 0, stream>>>(
      x, n1g, n1b, qkv_w, qkv_b, proj_w, proj_b, relb,
      n2g, n2b, fc1_w, fc1_b, fc2_w, fc2_b, out);
}
